// RVQTokenizer_46583215292834
// MI455X (gfx1250) — matrix operand order repaired and checked by reading, not compiled, not run on hardware
//
#include <hip/hip_runtime.h>

typedef __attribute__((ext_vector_type(16))) _Float16 v16h;
typedef __attribute__((ext_vector_type(8)))  _Float16 v8h;
typedef __attribute__((ext_vector_type(8)))  float    v8f;
typedef __attribute__((ext_vector_type(4)))  float    v4f;

#define RVQ_N 262144
#define RVQ_D 64
#define RVQ_Q 8
#define RVQ_K 1024

// Pre-pass: convert codebooks to f16 (for WMMA B fragments) and compute 0.5*||c||^2.
__global__ __launch_bounds__(256) void rvq_prep_f16(const float* __restrict__ cb,
                                                    _Float16* __restrict__ cb16,
                                                    float* __restrict__ hn) {
    int row = blockIdx.x * blockDim.x + threadIdx.x;   // 0 .. Q*K-1
    if (row >= RVQ_Q * RVQ_K) return;
    const float* src = cb + (size_t)row * RVQ_D;
    _Float16* dst = cb16 + (size_t)row * RVQ_D;
    float s = 0.f;
#pragma unroll
    for (int d = 0; d < RVQ_D; ++d) {
        float v = src[d];
        dst[d] = (_Float16)v;
        s += v * v;
    }
    hn[row] = 0.5f * s;
}

// Main: one wave per 16-row tile, 8 waves per block. Per stage, the block stages the
// full 128KB f16 stage-codebook (+4KB norms) in LDS once (CDNA5: 320KB LDS/workgroup),
// then each wave runs 64 code tiles of 2x v_wmma_f32_16x16x32_f16 with B fragments
// served by ds_load_b128. Residual kept in registers in A-fragment layout.
__global__ __launch_bounds__(256) void rvq_encode_wmma(const float* __restrict__ x,
                                                       const float* __restrict__ cbf,
                                                       const _Float16* __restrict__ cb16,
                                                       const float* __restrict__ hn,
                                                       float* __restrict__ out_enc,
                                                       float* __restrict__ out_q) {
    __shared__ __align__(32) _Float16 scb[RVQ_K * RVQ_D];  // 128 KB stage codebook
    __shared__ float shn[RVQ_K];                           //   4 KB half-norms

    const int lane = threadIdx.x & 31;
    const int wave = threadIdx.x >> 5;
    const int m    = lane & 15;   // row within 16-row tile
    const int hi   = lane >> 4;   // half-wave select
    const size_t rowTile = (size_t)blockIdx.x * 8 + wave;
    const size_t gm = rowTile * 16 + m;          // global row owned by this lane

    // This lane holds row gm's dims in 4 chunks of 8: chunk c -> dims c*16 + hi*8 .. +7
    float xr[32], rr[32];
#pragma unroll
    for (int c = 0; c < 4; ++c) {
        const int db = c * 16 + hi * 8;
        const v4f* p = (const v4f*)(x + gm * RVQ_D + db);
        v4f lo = p[0], hp = p[1];
#pragma unroll
        for (int e = 0; e < 4; ++e) { xr[c*8+e] = lo[e]; xr[c*8+4+e] = hp[e]; }
    }
#pragma unroll
    for (int e = 0; e < 32; ++e) rr[e] = xr[e];

    for (int q = 0; q < RVQ_Q; ++q) {
        // Fence previous stage's LDS readers, then stage this stage's codebook.
        __syncthreads();
        {
            const uint4* gsrc = (const uint4*)(cb16 + (size_t)q * RVQ_K * RVQ_D);
            uint4* ldst = (uint4*)scb;
#pragma unroll
            for (int i = threadIdx.x; i < (RVQ_K * RVQ_D) / 8; i += 256)  // 32 iters
                ldst[i] = gsrc[i];
            const float* hsrc = hn + (size_t)q * RVQ_K;
#pragma unroll
            for (int i = threadIdx.x; i < RVQ_K; i += 256)                // 4 iters
                shn[i] = hsrc[i];
        }
        __syncthreads();

        // Build A fragments (16x32 f16, two K-halves of D=64) from residual registers.
        v16h a0, a1;
#pragma unroll
        for (int e = 0; e < 16; ++e) { a0[e] = (_Float16)rr[e]; a1[e] = (_Float16)rr[16 + e]; }

        float best[8];
        int   bidx[8];
#pragma unroll
        for (int j = 0; j < 8; ++j) { best[j] = -3.0e38f; bidx[j] = 0; }

#pragma unroll 2
        for (int t = 0; t < 64; ++t) {
            const int code = (t << 4) + m;                    // this lane's column
            const _Float16* brow = scb + code * RVQ_D;
            const v8h* bq = (const v8h*)(brow + hi * 8);
            v16h b0 = __builtin_shufflevector(bq[0], bq[2], 0,1,2,3,4,5,6,7,8,9,10,11,12,13,14,15);  // K = hi*8 .. +7, 16 + hi*8 .. +7
            v16h b1 = __builtin_shufflevector(bq[4], bq[6], 0,1,2,3,4,5,6,7,8,9,10,11,12,13,14,15);  // K = 32 + hi*8 .. +7, 48 + hi*8 .. +7
            const float h = shn[code];
            v8f acc = {};
            acc = __builtin_amdgcn_wmma_f32_16x16x32_f16(false, a0, false, b0,
                                                         (short)0, acc, false, false);
            acc = __builtin_amdgcn_wmma_f32_16x16x32_f16(false, a1, false, b1,
                                                         (short)0, acc, false, false);
#pragma unroll
            for (int j = 0; j < 8; ++j) {                     // row = j + 8*hi
                const float s = acc[j] - h;                   // argmax s == argmin dist
                if (s > best[j]) { best[j] = s; bidx[j] = code; }
            }
        }

        // Reduce across 16 columns inside each half-wave (tie-break: smaller index).
#pragma unroll
        for (int mask = 1; mask <= 8; mask <<= 1) {
#pragma unroll
            for (int j = 0; j < 8; ++j) {
                const float os = __shfl_xor(best[j], mask, 32);
                const int   oi = __shfl_xor(bidx[j], mask, 32);
                if (os > best[j] || (os == best[j] && oi < bidx[j])) {
                    best[j] = os; bidx[j] = oi;
                }
            }
        }

        // Pick winner for this lane's row m (slot j = m&7, half = m>>3).
        const int j = m & 7;
        const int c01 = (j & 1) ? bidx[1] : bidx[0];
        const int c23 = (j & 1) ? bidx[3] : bidx[2];
        const int c45 = (j & 1) ? bidx[5] : bidx[4];
        const int c67 = (j & 1) ? bidx[7] : bidx[6];
        const int c03 = (j & 2) ? c23 : c01;
        const int c47 = (j & 2) ? c67 : c45;
        const int cand  = (j & 4) ? c47 : c03;
        const int other = __shfl_xor(cand, 16, 32);
        const int sel   = ((m >> 3) == hi) ? cand : other;

        if (hi == 0) out_enc[gm * RVQ_Q + q] = (float)sel;    // encoded[n, q]

        // Residual update with the f32 codebook row (global; 1 row per lane, L2-hot).
        const float* crow = cbf + ((size_t)q * RVQ_K + sel) * RVQ_D;
#pragma unroll
        for (int c = 0; c < 4; ++c) {
            const int db = c * 16 + hi * 8;
            const v4f* p = (const v4f*)(crow + db);
            v4f lo = p[0], hp = p[1];
#pragma unroll
            for (int e = 0; e < 4; ++e) { rr[c*8+e] -= lo[e]; rr[c*8+4+e] -= hp[e]; }
        }
    }

    // quantized = x - final residual
#pragma unroll
    for (int c = 0; c < 4; ++c) {
        const int db = c * 16 + hi * 8;
        v4f lo, hp;
#pragma unroll
        for (int e = 0; e < 4; ++e) {
            lo[e] = xr[c*8+e]   - rr[c*8+e];
            hp[e] = xr[c*8+4+e] - rr[c*8+4+e];
        }
        v4f* p = (v4f*)(out_q + gm * RVQ_D + db);
        p[0] = lo; p[1] = hp;
    }
}

extern "C" void kernel_launch(void* const* d_in, const int* in_sizes, int n_in,
                              void* d_out, int out_size, void* d_ws, size_t ws_size,
                              hipStream_t stream) {
    const float* x   = (const float*)d_in[0];        // [N, D] f32
    const float* cb  = (const float*)d_in[1];        // [Q, K, D] f32

    _Float16* cb16 = (_Float16*)d_ws;                                        // 1 MB
    float*    hn   = (float*)((char*)d_ws +
                              (size_t)RVQ_Q * RVQ_K * RVQ_D * sizeof(_Float16)); // 32 KB

    float* out_enc = (float*)d_out;                  // [N, Q] (indices as float)
    float* out_q   = out_enc + (size_t)RVQ_N * RVQ_Q; // [N, D] quantized

    rvq_prep_f16<<<(RVQ_Q * RVQ_K + 255) / 256, 256, 0, stream>>>(cb, cb16, hn);
    rvq_encode_wmma<<<RVQ_N / 128, 256, 0, stream>>>(x, cb, cb16, hn, out_enc, out_q);
}
